// VQEmbeddingEMA_DDP_83124797047447
// MI455X (gfx1250) — matrix operand order repaired and checked by reading, not compiled, not run on hardware
//
#include <hip/hip_runtime.h>
#include <hip/hip_bf16.h>

typedef __attribute__((ext_vector_type(16))) __bf16 v16bf;
typedef __attribute__((ext_vector_type(4)))  __bf16 v4bf;
typedef __attribute__((ext_vector_type(8)))  float  v8f;

#define Bsz 16
#define Tsz 4096
#define Dd  256
#define Mm  1024
#define Nn  (Bsz * Tsz)   // 65536 rows

__device__ __forceinline__ void bf_split(float f, __bf16& h, __bf16& l) {
    h = (__bf16)f;
    l = (__bf16)(f - (float)h);
}

// ---------------------------------------------------------------------------
// Kernel 1: one-time prep of the codebook (L2-resident, 1 MB):
//   emb_hi/emb_lo = hi/lo bf16 split (row-major, directly WMMA-B loadable)
//   enorm[j]      = sum_k emb[j][k]^2  (exact fp32)
// one wave per row, 4 waves per block
// ---------------------------------------------------------------------------
__global__ __launch_bounds__(128) void vq_prep(const float* __restrict__ emb,
                                               __bf16* __restrict__ emb_hi,
                                               __bf16* __restrict__ emb_lo,
                                               float* __restrict__ enorm) {
    const int lane = threadIdx.x & 31;
    const int wave = threadIdx.x >> 5;
    const int row  = blockIdx.x * 4 + wave;
    const float4* e4 = (const float4*)(emb + (size_t)row * Dd);
    __bf16* hrow = emb_hi + (size_t)row * Dd;
    __bf16* lrow = emb_lo + (size_t)row * Dd;

    float ss = 0.f;
    #pragma unroll
    for (int i = lane; i < Dd / 4; i += 32) {
        float4 v = e4[i];
        ss += v.x * v.x + v.y * v.y + v.z * v.z + v.w * v.w;
        float fl[4] = {v.x, v.y, v.z, v.w};
        v4bf h, l;
        #pragma unroll
        for (int e = 0; e < 4; ++e) {
            __bf16 hh, ll;
            bf_split(fl[e], hh, ll);
            h[e] = hh;
            l[e] = ll;
        }
        *(v4bf*)(hrow + i * 4) = h;
        *(v4bf*)(lrow + i * 4) = l;
    }
    #pragma unroll
    for (int off = 16; off > 0; off >>= 1) ss += __shfl_xor(ss, off, 32);
    if (lane == 0) enorm[row] = ss;
}

// ---------------------------------------------------------------------------
// Kernel 2: fused bf16x3-split WMMA GEMM (x @ e^T) + argmin over M
// 4 waves per block; each wave handles a 16-row tile across all M columns.
// B fragments stream pre-split from emb_hi/emb_lo: zero conversion VALU in
// the hot loop (2x 32B aligned loads -> B in the same k order as A).
// ---------------------------------------------------------------------------
__global__ __launch_bounds__(128) void vq_argmin(const float* __restrict__ x,
                                                 const __bf16* __restrict__ emb_hi,
                                                 const __bf16* __restrict__ emb_lo,
                                                 const float* __restrict__ enorm,
                                                 int* __restrict__ idx_i,
                                                 float* __restrict__ idx_f) {
    const int lane = threadIdx.x & 31;
    const int wave = threadIdx.x >> 5;
    const int lr   = lane & 15;   // row-in-tile for A / col-in-tile for B
    const int lh   = lane >> 4;   // K-half selector
    const int row0 = (blockIdx.x * 4 + wave) * 16;

    // ---- load A fragments for the full K=256 (8 steps of 32), split hi/lo
    // Layout used, not the ISA table order (16-bit A 16x32): lane(lh,lr) holds contiguous
    // K = 32s+16lh+[0,16), packed 2 per VGPR -> v16bf elements in order.
    v16bf a_hi[8], a_lo[8];
    const float* xrow = x + (size_t)(row0 + lr) * Dd;
    #pragma unroll
    for (int s = 0; s < 8; ++s) {
        const int kb = s * 32 + 16 * lh;
        float4 f0 = *(const float4*)(xrow + kb);
        float4 f1 = *(const float4*)(xrow + kb + 4);
        float4 f2 = *(const float4*)(xrow + kb + 8);
        float4 f3 = *(const float4*)(xrow + kb + 12);
        float fl[16] = {f0.x, f0.y, f0.z, f0.w, f1.x, f1.y, f1.z, f1.w,
                        f2.x, f2.y, f2.z, f2.w, f3.x, f3.y, f3.z, f3.w};
        #pragma unroll
        for (int e = 0; e < 16; ++e) {
            __bf16 h, l;
            bf_split(fl[e], h, l);
            a_hi[s][e] = h;
            a_lo[s][e] = l;
        }
    }

    float best[8];
    int   bidx[8];
    #pragma unroll
    for (int r = 0; r < 8; ++r) { best[r] = 3.4e38f; bidx[r] = 0; }

    for (int t = 0; t < Mm / 16; ++t) {
        const int col = t * 16 + lr;
        const __bf16* ehrow = emb_hi + (size_t)col * Dd;
        const __bf16* elrow = emb_lo + (size_t)col * Dd;
        v8f acc = {};
        #pragma unroll
        for (int s = 0; s < 8; ++s) {
            // Layout used, same k order as A (16-bit B 32x16): lane(lh,lr) holds contiguous
            // K = 32s+16lh+[0,16) of column lr -> one 32B load each.
            const int kb = s * 32 + 16 * lh;
            v16bf b_hi = *(const v16bf*)(ehrow + kb);
            v16bf b_lo = *(const v16bf*)(elrow + kb);
            // bf16x3 split product: xh*eh + xh*el + xl*eh  (~fp32 accurate)
            acc = __builtin_amdgcn_wmma_f32_16x16x32_bf16(
                false, a_hi[s], false, b_hi, (short)0, acc, false, false);
            acc = __builtin_amdgcn_wmma_f32_16x16x32_bf16(
                false, a_hi[s], false, b_lo, (short)0, acc, false, false);
            acc = __builtin_amdgcn_wmma_f32_16x16x32_bf16(
                false, a_lo[s], false, b_hi, (short)0, acc, false, false);
        }
        // C/D layout: lane(lh,lr) vgpr r -> dot(row0 + 8*lh + r, col)
        const float en = enorm[col];
        #pragma unroll
        for (int r = 0; r < 8; ++r) {
            float d = en - 2.0f * acc[r];  // + ||x||^2 is argmin-invariant
            if (d < best[r]) { best[r] = d; bidx[r] = col; }
        }
    }

    // reduce (min, first-idx) across the 16 lanes of each half-wave
    #pragma unroll
    for (int off = 1; off < 16; off <<= 1) {
        #pragma unroll
        for (int r = 0; r < 8; ++r) {
            float ob = __shfl_xor(best[r], off, 32);
            int   oi = __shfl_xor(bidx[r], off, 32);
            if (ob < best[r] || (ob == best[r] && oi < bidx[r])) {
                best[r] = ob;
                bidx[r] = oi;
            }
        }
    }
    if (lr == 0) {
        #pragma unroll
        for (int r = 0; r < 8; ++r) {
            const int row = row0 + 8 * lh + r;
            idx_i[row] = bidx[r];
            idx_f[row] = (float)bidx[r];
        }
    }
}

// ---------------------------------------------------------------------------
// Kernel 3: gather quantized rows, straight-through output, loss + counts
// 256 threads = 4 rows x 64 lanes (one float4 each)
// ---------------------------------------------------------------------------
__global__ __launch_bounds__(256) void vq_gather(const float* __restrict__ x,
                                                 const float* __restrict__ emb,
                                                 const int* __restrict__ idx,
                                                 float* __restrict__ qst,
                                                 unsigned int* __restrict__ counts,
                                                 float* __restrict__ loss_acc) {
    __shared__ float red[8];
    const int t  = threadIdx.x;
    const int rl = t >> 6;  // 0..3 row within block
    const int c  = t & 63;  // float4 slot within row
    const int row = blockIdx.x * 4 + rl;
    const int j   = idx[row];

    const float4 q4 = ((const float4*)(emb + (size_t)j * Dd))[c];
    const float4 x4 = ((const float4*)(x + (size_t)row * Dd))[c];

    float4 st;  // straight-through: x + (q - x), computed exactly as reference
    st.x = x4.x + (q4.x - x4.x);
    st.y = x4.y + (q4.y - x4.y);
    st.z = x4.z + (q4.z - x4.z);
    st.w = x4.w + (q4.w - x4.w);
    ((float4*)(qst + (size_t)row * Dd))[c] = st;

    float dx = x4.x - q4.x, dy = x4.y - q4.y, dz = x4.z - q4.z, dw = x4.w - q4.w;
    float ss = dx * dx + dy * dy + dz * dz + dw * dw;
    #pragma unroll
    for (int off = 16; off > 0; off >>= 1) ss += __shfl_xor(ss, off, 32);
    if ((t & 31) == 0) red[t >> 5] = ss;
    __syncthreads();
    if (t < 8) {
        float v = red[t];
        #pragma unroll
        for (int off = 4; off > 0; off >>= 1) v += __shfl_xor(v, off, 32);
        if (t == 0) atomicAdd(loss_acc, v);
    }
    if (c == 0) atomicAdd(&counts[j], 1u);
}

// ---------------------------------------------------------------------------
// Kernel 4: usage stats (single block, M=1024 threads)
// ---------------------------------------------------------------------------
__global__ __launch_bounds__(1024) void vq_stats(const unsigned int* __restrict__ counts,
                                                 const float* __restrict__ loss_acc,
                                                 float* __restrict__ scalars) {
    __shared__ float s_used[32], s_ent[32];
    const int t = threadIdx.x;
    const int lane = t & 31, wid = t >> 5;
    float cnt  = (float)counts[t];
    float used = (cnt >= 1.0f) ? 1.0f : 0.0f;
    float avg  = cnt / (float)Nn;
    float ent  = avg * logf(avg + 1e-10f);
    #pragma unroll
    for (int off = 16; off > 0; off >>= 1) {
        used += __shfl_xor(used, off, 32);
        ent  += __shfl_xor(ent, off, 32);
    }
    if (lane == 0) { s_used[wid] = used; s_ent[wid] = ent; }
    __syncthreads();
    if (t < 32) {
        float u = s_used[t], e = s_ent[t];
        #pragma unroll
        for (int off = 16; off > 0; off >>= 1) {
            u += __shfl_xor(u, off, 32);
            e += __shfl_xor(e, off, 32);
        }
        if (t == 0) {
            scalars[0] = u;                                   // used_curr
            scalars[1] = expf(-e);                            // perplexity
            scalars[2] = loss_acc[0] / (float)(Nn * Dd);      // commitment loss
        }
    }
}

// ---------------------------------------------------------------------------
extern "C" void kernel_launch(void* const* d_in, const int* in_sizes, int n_in,
                              void* d_out, int out_size, void* d_ws, size_t ws_size,
                              hipStream_t stream) {
    const float* x   = (const float*)d_in[0];   // [16,4096,256]
    const float* emb = (const float*)d_in[1];   // [1024,256]

    float* out     = (float*)d_out;
    float* qst     = out;                         // [N, D]
    float* idx_f   = out + (size_t)Nn * Dd;       // [N]
    float* scalars = idx_f + Nn;                  // used, ppl, loss

    char* ws = (char*)d_ws;
    int*          idx_i    = (int*)ws;                             // 256 KB
    float*        enorm    = (float*)(ws + 262144);                // 4 KB
    unsigned int* counts   = (unsigned int*)(ws + 266240);         // 4 KB
    float*        loss_acc = (float*)(ws + 270336);                // 256 B
    __bf16*       emb_hi   = (__bf16*)(ws + 270592);               // 512 KB
    __bf16*       emb_lo   = (__bf16*)(ws + 794880);               // 512 KB

    (void)hipMemsetAsync(counts, 0, 4096 + 256, stream);  // zero counts + loss

    vq_prep  <<<Mm / 4, 128, 0, stream>>>(emb, emb_hi, emb_lo, enorm);
    vq_argmin<<<Nn / 64, 128, 0, stream>>>(x, emb_hi, emb_lo, enorm, idx_i, idx_f);
    vq_gather<<<Nn / 4, 256, 0, stream>>>(x, emb, idx_i, qst, counts, loss_acc);
    vq_stats <<<1, 1024, 0, stream>>>(counts, loss_acc, scalars);
}
